// HeteroGNN_44298292691035
// MI455X (gfx1250) — matrix operand order repaired and checked by reading, not compiled, not run on hardware
//
#include <hip/hip_runtime.h>
#include <hip/hip_bf16.h>
#include <cstdint>
#include <cstddef>

// ---------------- problem constants ----------------
#define NP   100000
#define NF   100000
#define NPO  50000
#define NT   (NP + NF + NPO)          // 250000 nodes total, order: process|file|port
#define EDG  200000
#define HDIM 128
#define GG   128
#define OUTD 2

#define OFF_P  0
#define OFF_F  NP
#define OFF_PO (NP + NF)

typedef __attribute__((ext_vector_type(16))) __bf16 v16bf;
typedef __attribute__((ext_vector_type(8)))  float  v8f;

// ---------------- helpers ----------------
__device__ __forceinline__ unsigned f2bf_pair(float a, float b) {
  unsigned ua = __float_as_uint(a);
  unsigned ub = __float_as_uint(b);
  ua = (ua + 0x7FFFu + ((ua >> 16) & 1u)) >> 16;   // RNE f32 -> bf16
  ub = (ub + 0x7FFFu + ((ub >> 16) & 1u)) >> 16;
  return (ua & 0xFFFFu) | (ub << 16);
}

__device__ __forceinline__ unsigned float_order_key(float f) {
  unsigned u = __float_as_uint(f);
  return (u & 0x80000000u) ? ~u : (u | 0x80000000u);   // monotonic uint key
}
__device__ __forceinline__ float float_order_unkey(unsigned k) {
  unsigned u = (k & 0x80000000u) ? (k & 0x7FFFFFFFu) : ~k;
  return __uint_as_float(u);
}

// ---------------- feature conversion (layer 0 inputs -> packed bf16 pairs) ----------------
__global__ void k_convert_inputs(const float* __restrict__ xp,
                                 const float* __restrict__ xf,
                                 const float* __restrict__ xpo,
                                 unsigned* __restrict__ xb /* NT x 32 uint (K=64) */) {
  int i = blockIdx.x * blockDim.x + threadIdx.x;
  if (i >= NT * 32) return;
  int node = i >> 5;          // /32 pairs per node
  int p    = i & 31;
  const float* src;
  if (node < NP)            src = xp  + (size_t)node * 64;
  else if (node < NP + NF)  src = xf  + (size_t)(node - NP) * 64;
  else                      src = xpo + (size_t)(node - NP - NF) * 64;
  xb[(size_t)node * 32 + p] = f2bf_pair(src[2 * p], src[2 * p + 1]);
}

// ---------------- weight conversion: W (KxN f32) -> per-lane WMMA B-fragment order ----------------
// wb[((k0/32)*8 + nt)*256 + lane*8 + v] = pack(B[kk][col], B[kk+1][col])
//   kk  = k0 + 2*v + 16*(lane>>4), col = nt*16 + (lane&15)   (contiguous k order, same as the A fragment)
// -> each lane's 8 fragment dwords are contiguous: two ds_load_b128 per fragment.
__global__ void k_convert_w_sw(const float* __restrict__ W, unsigned* __restrict__ wb, int K) {
  int i = blockIdx.x * blockDim.x + threadIdx.x;
  int total = (K >> 5) * 2048;                 // (K/32) ksteps * 8 ntiles * 256 lanes*v
  if (i >= total) return;
  int ks   = i >> 11;
  int rem  = i & 2047;
  int nt   = rem >> 8;
  int li   = rem & 255;
  int lane = li >> 3, v = li & 7;
  int kk   = ks * 32 + 2 * v + 16 * (lane >> 4);
  int col  = nt * 16 + (lane & 15);
  wb[i] = f2bf_pair(W[(size_t)kk * HDIM + col], W[(size_t)(kk + 1) * HDIM + col]);
}

// ---------------- WMMA GEMM: C[M x 128] = A[M x K](bf16) * B[K x 128](bf16), f32 accum ----------------
template<int K>
__global__ __launch_bounds__(256)
void k_gemm(const unsigned* __restrict__ A /* M x K/2 uint (bf16 pairs, row major) */,
            const unsigned* __restrict__ Bsw /* (K/32)*2048 uint, fragment-swizzled */,
            float* __restrict__ C, int M) {
  constexpr int KH = K / 2;
  __shared__ unsigned sA[128 * KH];
  __shared__ unsigned sB[(K / 32) * 2048];

  const int tid  = threadIdx.x;
  const int lane = tid & 31;
  const int wave = tid >> 5;
  const int rowBase = blockIdx.x * 128;

  for (int i = tid; i < (K / 32) * 2048; i += 256) sB[i] = Bsw[i];
  for (int i = tid; i < 128 * KH; i += 256) {
    int r = i / KH, c = i - r * KH;
    int gr = rowBase + r; if (gr >= M) gr = M - 1;
    sA[i] = A[(size_t)gr * KH + c];
  }
  __syncthreads();

  const int m0   = wave * 16;
  const int l15  = lane & 15;
  const int grp  = lane >> 4;          // 0: lanes 0-15, 1: lanes 16-31
  const int rowA = m0 + l15;

  v8f accv[8];
  #pragma unroll
  for (int nt = 0; nt < 8; ++nt) accv[nt] = v8f{0, 0, 0, 0, 0, 0, 0, 0};

  #pragma unroll
  for (int k0 = 0; k0 < K; k0 += 32) {
    // A fragment (contiguous k order, as B): 8 contiguous pair-dwords -> 2x ds_load_b128
    union { v16bf v; uint4 q[2]; } af;
    const unsigned* aptr = &sA[rowA * KH + (k0 >> 1) + 8 * grp];
    af.q[0] = *(const uint4*)(aptr);
    af.q[1] = *(const uint4*)(aptr + 4);

    const int ksBase = (k0 >> 5) * 2048 + lane * 8;
    #pragma unroll
    for (int nt = 0; nt < 8; ++nt) {
      union { v16bf v; uint4 q[2]; } bfv;
      const unsigned* bptr = &sB[ksBase + nt * 256];
      bfv.q[0] = *(const uint4*)(bptr);
      bfv.q[1] = *(const uint4*)(bptr + 4);
      accv[nt] = __builtin_amdgcn_wmma_f32_16x16x32_bf16(
          false, af.v, false, bfv.v, (short)0, accv[nt], false, false);
    }
  }

  // D layout: VGPR r -> row m0 + r + 8*grp, col = nt*16 + (lane&15)
  const int rbase = rowBase + m0 + 8 * grp;
  if (rowBase + 128 <= M) {
    #pragma unroll
    for (int nt = 0; nt < 8; ++nt) {
      const int col = nt * 16 + l15;
      #pragma unroll
      for (int r = 0; r < 8; ++r)
        C[(size_t)(rbase + r) * HDIM + col] = accv[nt][r];
    }
  } else {
    #pragma unroll
    for (int nt = 0; nt < 8; ++nt) {
      const int col = nt * 16 + l15;
      #pragma unroll
      for (int r = 0; r < 8; ++r)
        if (rbase + r < M) C[(size_t)(rbase + r) * HDIM + col] = accv[nt][r];
    }
  }
}

// ---------------- per-node attention scalars: s = h . a_s, d = h . a_d ----------------
__global__ __launch_bounds__(256)
void k_sd(const float* __restrict__ h, const float* __restrict__ as,
          const float* __restrict__ ad, float* __restrict__ s, float* __restrict__ d, int n) {
  int gid  = blockIdx.x * blockDim.x + threadIdx.x;
  int node = gid >> 5;
  int lane = threadIdx.x & 31;
  if (node >= n) return;
  const float* row = h + (size_t)node * HDIM;
  float ss = 0.f, dd = 0.f;
  #pragma unroll
  for (int j = lane; j < HDIM; j += 32) { float v = row[j]; ss += v * as[j]; dd += v * ad[j]; }
  #pragma unroll
  for (int o = 16; o > 0; o >>= 1) { ss += __shfl_down(ss, o, 32); dd += __shfl_down(dd, o, 32); }
  if (lane == 0) { s[node] = ss; d[node] = dd; }
}

// ---------------- relation phase 1: alpha + segment max (ordered-uint atomicMax) ----------------
__global__ void k_alpha(const int* __restrict__ ei, const float* __restrict__ s,
                        const float* __restrict__ d, int so, int dof,
                        float* __restrict__ alpha, unsigned* __restrict__ mu) {
  int e = blockIdx.x * blockDim.x + threadIdx.x;
  if (e >= EDG) return;
  int es = ei[e], ed = ei[EDG + e];
  float a  = s[so + es] + d[dof + ed];
  float al = a > 0.f ? a : 0.2f * a;          // leaky_relu(., 0.2)
  alpha[e] = al;
  atomicMax(&mu[ed], float_order_key(al));
}

// ---------------- relation phase 2: exp(alpha - max) + denominator ----------------
__global__ void k_ex(const int* __restrict__ ei, float* __restrict__ alpha,
                     const unsigned* __restrict__ mu, float* __restrict__ den) {
  int e = blockIdx.x * blockDim.x + threadIdx.x;
  if (e >= EDG) return;
  int ed = ei[EDG + e];
  float mm = float_order_unkey(mu[ed]);
  float ex = __expf(alpha[e] - mm);
  alpha[e] = ex;
  unsafeAtomicAdd(&den[ed], ex);
}

// ---------------- relation phase 3: weighted scatter-add (1 wave per edge, 4 f32/lane) ----------------
__global__ __launch_bounds__(256)
void k_scatter(const int* __restrict__ ei, const float* __restrict__ alpha,
               const float* __restrict__ den, const float* __restrict__ h,
               int so, int dof, float* __restrict__ acc) {
  int gid  = blockIdx.x * blockDim.x + threadIdx.x;
  int e    = gid >> 5;
  int lane = threadIdx.x & 31;
  if (e >= EDG) return;
  int es = ei[e], ed = ei[EDG + e];
  float w = alpha[e] / fmaxf(den[ed], 1e-16f);
  const float4 v = ((const float4*)(h + (size_t)(so + es) * HDIM))[lane];
  float* dst = acc + (size_t)(dof + ed) * HDIM + lane * 4;
  unsafeAtomicAdd(dst + 0, w * v.x);
  unsafeAtomicAdd(dst + 1, w * v.y);
  unsafeAtomicAdd(dst + 2, w * v.z);
  unsafeAtomicAdd(dst + 3, w * v.w);
}

// ---------------- mean over relations + bias + leaky_relu(0.01); write f32 (in place) + bf16 pairs ----------------
__global__ void k_finish(float* __restrict__ acc, unsigned* __restrict__ xb /* NT x 64 uint */,
                         const float* __restrict__ b, float inv, int off, int n) {
  int i = blockIdx.x * blockDim.x + threadIdx.x;
  if (i >= n * 64) return;
  int node = off + i / 64;
  int p    = i % 64;
  size_t base = (size_t)node * HDIM + 2 * p;
  float v0 = acc[base]     * inv + b[2 * p];
  float v1 = acc[base + 1] * inv + b[2 * p + 1];
  v0 = v0 > 0.f ? v0 : 0.01f * v0;
  v1 = v1 > 0.f ? v1 : 0.01f * v1;
  acc[base] = v0; acc[base + 1] = v1;
  xb[(size_t)node * 64 + p] = f2bf_pair(v0, v1);
}

// ---------------- global add pool ----------------
__global__ void k_pool(const float* __restrict__ x, const int* __restrict__ batch,
                       float* __restrict__ pooled, int off, int n) {
  int i = blockIdx.x * blockDim.x + threadIdx.x;
  if (i >= n * HDIM) return;
  int node = i / HDIM, c = i % HDIM;
  int g = batch[node];
  unsafeAtomicAdd(&pooled[(size_t)g * HDIM + c], x[(size_t)(off + node) * HDIM + c]);
}

// ---------------- head linears (tiny) ----------------
__global__ void k_lin1(const float* __restrict__ pooled, const float* __restrict__ w,
                       const float* __restrict__ b, float* __restrict__ h1) {
  int i = blockIdx.x * blockDim.x + threadIdx.x;
  if (i >= GG * HDIM) return;
  int g = i / HDIM, j = i % HDIM;
  float acc = b[j];
  for (int c = 0; c < HDIM; ++c) acc += pooled[(size_t)g * HDIM + c] * w[(size_t)c * HDIM + j];
  h1[i] = acc;
}

__global__ void k_lin2(const float* __restrict__ h1, const float* __restrict__ w,
                       const float* __restrict__ b, float* __restrict__ out) {
  int i = blockIdx.x * blockDim.x + threadIdx.x;
  if (i >= GG * OUTD) return;
  int g = i / OUTD, o = i % OUTD;
  float acc = b[o];
  for (int j = 0; j < HDIM; ++j) acc += h1[(size_t)g * HDIM + j] * w[(size_t)j * OUTD + o];
  out[i] = acc;
}

// ---------------- host driver ----------------
static inline int cdiv(long a, long b) { return (int)((a + b - 1) / b); }

extern "C" void kernel_launch(void* const* d_in, const int* in_sizes, int n_in,
                              void* d_out, int out_size, void* d_ws, size_t ws_size,
                              hipStream_t stream) {
  (void)in_sizes; (void)n_in; (void)out_size; (void)ws_size;

  // inputs
  const float* xp  = (const float*)d_in[0];
  const float* xf  = (const float*)d_in[1];
  const float* xpo = (const float*)d_in[2];
  const float* Wl[3]  = {(const float*)d_in[3], (const float*)d_in[7],  (const float*)d_in[11]};
  const float* asl[3] = {(const float*)d_in[4], (const float*)d_in[8],  (const float*)d_in[12]};
  const float* adl[3] = {(const float*)d_in[5], (const float*)d_in[9],  (const float*)d_in[13]};
  const float* bl[3]  = {(const float*)d_in[6], (const float*)d_in[10], (const float*)d_in[14]};
  const float* lin1_w = (const float*)d_in[15];
  const float* lin1_b = (const float*)d_in[16];
  const float* lin2_w = (const float*)d_in[17];
  const float* lin2_b = (const float*)d_in[18];
  const int* batch_p  = (const int*)d_in[29];
  const int* batch_f  = (const int*)d_in[30];
  const int* batch_po = (const int*)d_in[31];
  float* out = (float*)d_out;

  // workspace partition
  size_t off = 0;
  auto wsa = [&](size_t bytes) -> void* {
    void* p = (char*)d_ws + off;
    off += (bytes + 255) & ~(size_t)255;
    return p;
  };
  float*    h      = (float*)   wsa((size_t)NT * HDIM * 4);   // dense transform output
  float*    acc    = (float*)   wsa((size_t)NT * HDIM * 4);   // relation accumulator / final x
  unsigned* xb     = (unsigned*)wsa((size_t)NT * 64 * 4);     // packed bf16 features (max K=128)
  unsigned* wb     = (unsigned*)wsa((size_t)64 * HDIM * 4);   // swizzled bf16 weights
  float*    sbuf   = (float*)   wsa((size_t)NT * 4);
  float*    dbuf   = (float*)   wsa((size_t)NT * 4);
  float*    alpha  = (float*)   wsa((size_t)EDG * 4);
  unsigned* mu     = (unsigned*)wsa((size_t)NP * 4);          // max n_dst = 100000
  float*    den    = (float*)   wsa((size_t)NP * 4);
  float*    pooled = (float*)   wsa((size_t)GG * HDIM * 4);
  float*    h1     = (float*)   wsa((size_t)GG * HDIM * 4);

  // relation table: {edge-index input, src offset, dst offset, n_dst}
  struct Rel { int ei; int so; int dof; int ndst; };
  static const Rel rels[10] = {
    {20, OFF_F,  OFF_P,  NP },   // rev_access : file -> process
    {22, OFF_F,  OFF_P,  NP },   // rev_same_as: file -> process
    {24, OFF_PO, OFF_P,  NP },   // rev_bind   : port -> process
    {27, OFF_P,  OFF_P,  NP },   // create     : process -> process
    {28, OFF_P,  OFF_P,  NP },   // rev_create : process -> process
    {19, OFF_P,  OFF_F,  NF },   // access     : process -> file
    {21, OFF_P,  OFF_F,  NF },   // same_as    : process -> file
    {23, OFF_P,  OFF_PO, NPO},   // bind       : process -> port
    {25, OFF_PO, OFF_PO, NPO},   // session    : port -> port
    {26, OFF_PO, OFF_PO, NPO},   // rev_session: port -> port
  };

  // layer 0 input features -> packed bf16 (K = 64)
  k_convert_inputs<<<cdiv((long)NT * 32, 256), 256, 0, stream>>>(xp, xf, xpo, xb);

  for (int l = 0; l < 3; ++l) {
    const int K = (l == 0) ? 64 : 128;

    k_convert_w_sw<<<cdiv((long)(K >> 5) * 2048, 256), 256, 0, stream>>>(Wl[l], wb, K);

    if (l == 0)
      k_gemm<64><<<cdiv(NT, 128), 256, 0, stream>>>(xb, wb, h, NT);
    else
      k_gemm<128><<<cdiv(NT, 128), 256, 0, stream>>>(xb, wb, h, NT);

    k_sd<<<cdiv((long)NT * 32, 256), 256, 0, stream>>>(h, asl[l], adl[l], sbuf, dbuf, NT);

    hipMemsetAsync(acc, 0, (size_t)NT * HDIM * 4, stream);

    for (int r = 0; r < 10; ++r) {
      const Rel& R = rels[r];
      const int* ei = (const int*)d_in[R.ei];
      hipMemsetAsync(mu,  0, (size_t)R.ndst * 4, stream);   // key 0 == -"infinity"
      hipMemsetAsync(den, 0, (size_t)R.ndst * 4, stream);
      k_alpha  <<<cdiv(EDG, 256), 256, 0, stream>>>(ei, sbuf, dbuf, R.so, R.dof, alpha, mu);
      k_ex     <<<cdiv(EDG, 256), 256, 0, stream>>>(ei, alpha, mu, den);
      k_scatter<<<cdiv((long)EDG * 32, 256), 256, 0, stream>>>(ei, alpha, den, h, R.so, R.dof, acc);
    }

    // mean over relations (+bias folded) + leaky_relu(0.01) -> f32 in acc, bf16 pairs in xb
    k_finish<<<cdiv((long)NP  * 64, 256), 256, 0, stream>>>(acc, xb, bl[l], 1.f / 5.f, OFF_P,  NP);
    k_finish<<<cdiv((long)NF  * 64, 256), 256, 0, stream>>>(acc, xb, bl[l], 1.f / 2.f, OFF_F,  NF);
    k_finish<<<cdiv((long)NPO * 64, 256), 256, 0, stream>>>(acc, xb, bl[l], 1.f / 3.f, OFF_PO, NPO);
  }

  // global add pool over all nodes (final x lives in acc)
  hipMemsetAsync(pooled, 0, (size_t)GG * HDIM * 4, stream);
  k_pool<<<cdiv((long)NP  * HDIM, 256), 256, 0, stream>>>(acc, batch_p,  pooled, OFF_P,  NP);
  k_pool<<<cdiv((long)NF  * HDIM, 256), 256, 0, stream>>>(acc, batch_f,  pooled, OFF_F,  NF);
  k_pool<<<cdiv((long)NPO * HDIM, 256), 256, 0, stream>>>(acc, batch_po, pooled, OFF_PO, NPO);

  k_lin1<<<cdiv(GG * HDIM, 256), 256, 0, stream>>>(pooled, lin1_w, lin1_b, h1);
  k_lin2<<<1, 256, 0, stream>>>(h1, lin2_w, lin2_b, out);
}
